// BailingMoeV2GroupedExperts_61495341744590
// MI455X (gfx1250) — matrix operand order repaired and checked by reading, not compiled, not run on hardware
//
#include <hip/hip_runtime.h>
#include <hip/hip_bf16.h>

// ---------------- problem constants (match reference) ----------------
constexpr int BB = 4, SS = 4096, HH = 2048, II = 1024, EE = 32, KK = 4;
constexpr int NN = BB * SS;              // 16384 tokens
constexpr int AT = NN * KK;              // 65536 assignments
constexpr int TILE_M = 32;               // rows per expert tile (2 wave-rows)
constexpr int MAX_TILES = AT / TILE_M + EE;  // 2080 upper bound
constexpr int HSTRIDE = II + 16;         // LDS hmid row stride (bf16 elems), 16B aligned
constexpr int BSTR = 40;                 // LDS stage row stride (bf16 elems), 16B aligned

typedef __attribute__((ext_vector_type(16))) __bf16 v16bf;
typedef __attribute__((ext_vector_type(8)))  float  v8f;

// fp32 -> bf16 round-to-nearest-even
__device__ __forceinline__ __bf16 f2bf(float f) {
  union { float f; unsigned u; } a; a.f = f;
  unsigned r = a.u + 0x7FFFu + ((a.u >> 16) & 1u);
  union { unsigned short s; __bf16 h; } b; b.s = (unsigned short)(r >> 16);
  return b.h;
}

// load a 16-element bf16 fragment as two aligned 16B loads (LDS or global)
__device__ __forceinline__ v16bf ld_frag2(const __bf16* p, int off2) {
  union { v16bf v; uint4 q[2]; } u;
  u.q[0] = *(const uint4*)(p);
  u.q[1] = *(const uint4*)(p + off2);
  return u.v;
}

// ---------------- kernel 1: zero output + routing meta ----------------
__global__ void moe_zero_kernel(float* __restrict__ out, size_t n,
                                int* __restrict__ meta, int nmeta) {
  size_t i = (size_t)blockIdx.x * blockDim.x + threadIdx.x;
  size_t stride = (size_t)gridDim.x * blockDim.x;
  for (size_t j = i; j < n; j += stride) out[j] = 0.0f;
  if (i < (size_t)nmeta) meta[i] = 0;
}

// ---------------- kernel 2: per-expert histogram ----------------
__global__ void moe_count_kernel(const int* __restrict__ topk_idx,
                                 int* __restrict__ counts) {
  int a = blockIdx.x * blockDim.x + threadIdx.x;
  if (a < AT) atomicAdd(&counts[topk_idx[a]], 1);
}

// ---------------- kernel 3: scan + tile worklist ----------------
__global__ void moe_scan_kernel(const int* __restrict__ counts,
                                int* __restrict__ offsets,
                                int* __restrict__ ntiles,
                                int* __restrict__ tile_e,
                                int* __restrict__ tile_r) {
  if (threadIdx.x == 0 && blockIdx.x == 0) {
    int off = 0, t = 0;
    for (int e = 0; e < EE; ++e) {
      offsets[e] = off;
      int c = counts[e];
      int nt = (c + TILE_M - 1) / TILE_M;
      for (int i = 0; i < nt && t < MAX_TILES; ++i) { tile_e[t] = e; tile_r[t] = i; ++t; }
      off += c;
    }
    offsets[EE] = off;
    *ntiles = t;
  }
}

// ---------------- kernel 4: scatter assignments expert-sorted ----------------
__global__ void moe_scatter_kernel(const int* __restrict__ topk_idx,
                                   const float* __restrict__ topk_w,
                                   const int* __restrict__ offsets,
                                   int* __restrict__ cursors,
                                   int* __restrict__ tok_list,
                                   float* __restrict__ w_list) {
  int a = blockIdx.x * blockDim.x + threadIdx.x;
  if (a < AT) {
    int e = topk_idx[a];
    int pos = offsets[e] + atomicAdd(&cursors[e], 1);
    tok_list[pos] = a / KK;
    w_list[pos]  = topk_w[a];
  }
}

// ---------------- kernel 5a: x fp32 -> bf16 row-major ----------------
__global__ void moe_cvt_x_kernel(const float* __restrict__ x,
                                 __bf16* __restrict__ xbf, size_t n) {
  size_t i = (size_t)blockIdx.x * blockDim.x + threadIdx.x;
  size_t stride = (size_t)gridDim.x * blockDim.x;
  for (size_t j = i; j < n; j += stride) xbf[j] = f2bf(x[j]);
}

// ---------------- kernel 5b: pack weights into WMMA-B fragment tiles -------
// Tile = 64 n x 32 k bf16, flattened as (wn*32 + lane)*16 + h where
//   n = n0 + wn*16 + (lane&15),  k = k0 + ((lane<16)?0:16) + h
// gate/up: tiles indexed ((e*16 + t_n)*64 + t_k), n over I, k over H
// down   : tiles indexed ((e*32 + t_n)*32 + t_k), n over H, k over I
__global__ __launch_bounds__(256)
void moe_pack_kernel(const float* __restrict__ Wg,
                     const float* __restrict__ Wu,
                     const float* __restrict__ Wd,
                     __bf16* __restrict__ pkg,
                     __bf16* __restrict__ pku,
                     __bf16* __restrict__ pkd) {
  int bank = blockIdx.y;                 // 0=gate 1=up 2=down
  int bx   = blockIdx.x;                 // 32768 tiles per bank
  const float* src; __bf16* dst;
  int t_k, t_n, e, kdim;
  if (bank < 2) {                        // [I, H]: t_n over I/64, t_k over H/32
    t_k = bx & 63; t_n = (bx >> 6) & 15; e = bx >> 10; kdim = HH;
    src = (bank == 0) ? Wg : Wu;
    dst = (bank == 0) ? pkg : pku;
  } else {                               // [H, I]: t_n over H/64, t_k over I/32
    t_k = bx & 31; t_n = (bx >> 5) & 31; e = bx >> 10; kdim = II;
    src = Wd; dst = pkd;
  }
  const float* sbase = src + (size_t)e * II * HH;   // II*HH == HH*II elements
  __bf16* tbase = dst + (size_t)bx * 2048;
  int n0 = t_n * 64, k0 = t_k * 32;

  int tid = threadIdx.x;
  int h0 = (tid & 1) * 8;                // dstIdx = tid*8 + j ; h = h0 + j
  int L  = (tid >> 1) & 31;
  int wn = tid >> 6;
  int n  = n0 + wn * 16 + (L & 15);
  int kb = (L < 16) ? 0 : 16;
  const float* s = sbase + (size_t)n * kdim + k0 + kb + h0;
#pragma unroll
  for (int j = 0; j < 8; ++j) tbase[tid * 8 + j] = f2bf(s[j]);
}

// ---------------- kernel 6: fast expert tile (packed bf16, barrier-free) ---
__global__ __launch_bounds__(256)
void moe_expert_fast(const __bf16* __restrict__ xbf,
                     const __bf16* __restrict__ pkg,
                     const __bf16* __restrict__ pku,
                     const __bf16* __restrict__ pkd,
                     const int*   __restrict__ tok_list,
                     const float* __restrict__ w_list,
                     const int*   __restrict__ counts,
                     const int*   __restrict__ offsets,
                     const int*   __restrict__ ntiles,
                     const int*   __restrict__ tile_e,
                     const int*   __restrict__ tile_r,
                     float* __restrict__ out) {
  __shared__ __bf16 s_hmid[TILE_M * HSTRIDE];   // 32 x 1024 bf16, ~65 KB
  __shared__ int    s_tok[TILE_M];
  __shared__ float  s_w  [TILE_M];

  int bid = blockIdx.x;
  if (bid >= *ntiles) return;
  int e    = tile_e[bid];
  int trow = tile_r[bid];
  int base = offsets[e] + trow * TILE_M;
  int rows = min(TILE_M, counts[e] - trow * TILE_M);

  int tid  = threadIdx.x;
  int lane = tid & 31;
  int wave = tid >> 5;
  int wm   = wave >> 2;                  // 0..1 M sub-tile
  int wn   = wave & 3;                   // 0..3 N sub-tile

  if (tid < TILE_M) {
    int idx = base + min(tid, rows - 1);
    s_tok[tid] = tok_list[idx];
    s_w[tid]   = (tid < rows) ? w_list[idx] : 0.0f;
  }
  __syncthreads();

  const int m    = lane & 15;
  const int kba  = (lane < 16) ? 0 : 16;  // A-frag K base
  const int loff = (wn * 32 + lane) * 16; // packed-B lane offset (bf16 elems)
  const int myTok = s_tok[wm * 16 + m];
  const __bf16* xrow0 = xbf + (size_t)myTok * HH + kba;

  // -------- phase 1: hmid = silu(x Wg^T) * (x Wu^T) --------
  for (int tn = 0; tn < II / 64; ++tn) {
    v8f accg = {0.f,0.f,0.f,0.f,0.f,0.f,0.f,0.f};
    v8f accu = {0.f,0.f,0.f,0.f,0.f,0.f,0.f,0.f};
    const __bf16* gt = pkg + ((size_t)(e * 16 + tn) * 64) * 2048 + loff;
    const __bf16* ut = pku + ((size_t)(e * 16 + tn) * 64) * 2048 + loff;
#pragma unroll 4
    for (int tk = 0; tk < HH / 32; ++tk) {
      __builtin_prefetch((const void*)(gt + 2 * 2048), 0, 0);
      __builtin_prefetch((const void*)(ut + 2 * 2048), 0, 0);
      v16bf af = ld_frag2(xrow0 + tk * 32, 8);      // A direct from global bf16
      v16bf bg = ld_frag2(gt, 8);                   // B frags: packed, coalesced
      v16bf bu = ld_frag2(ut, 8);
      accg = __builtin_amdgcn_wmma_f32_16x16x32_bf16(false, af, false, bg,
                                                     (short)0, accg, false, false);
      accu = __builtin_amdgcn_wmma_f32_16x16x32_bf16(false, af, false, bu,
                                                     (short)0, accu, false, false);
      gt += 2048; ut += 2048;
    }
    { // silu(g)*u -> LDS hmid
      int mrow = (lane < 16) ? 0 : 8;
#pragma unroll
      for (int r = 0; r < 8; ++r) {
        float g   = accg[r];
        float u   = accu[r];
        float sig = __builtin_amdgcn_rcpf(1.0f + __expf(-g));
        float hm  = g * sig * u;
        s_hmid[(wm * 16 + mrow + r) * HSTRIDE + tn * 64 + wn * 16 + m] = f2bf(hm);
      }
    }
  }
  __syncthreads();   // hmid complete; phases 2 is barrier-free

  // -------- phase 2: o = hmid Wd^T, weighted atomic combine --------
  const __bf16* hrow = &s_hmid[(wm * 16 + m) * HSTRIDE + kba];
  for (int tn = 0; tn < HH / 64; ++tn) {
    v8f acc = {0.f,0.f,0.f,0.f,0.f,0.f,0.f,0.f};
    const __bf16* dt = pkd + ((size_t)(e * 32 + tn) * 32) * 2048 + loff;
#pragma unroll 4
    for (int tk = 0; tk < II / 32; ++tk) {
      __builtin_prefetch((const void*)(dt + 2 * 2048), 0, 0);
      v16bf af = ld_frag2(hrow + tk * 32, 8);
      v16bf bd = ld_frag2(dt, 8);
      acc = __builtin_amdgcn_wmma_f32_16x16x32_bf16(false, af, false, bd,
                                                    (short)0, acc, false, false);
      dt += 2048;
    }
    {
      int mrow = (lane < 16) ? 0 : 8;
      int col  = tn * 64 + wn * 16 + m;
#pragma unroll
      for (int r = 0; r < 8; ++r) {
        int row   = wm * 16 + mrow + r;
        float val = acc[r] * s_w[row];
        atomicAdd(&out[(size_t)s_tok[row] * HH + col], val);
      }
    }
  }
}

// ---------------- kernel 7: fallback fused expert tile (async LDS path) ----
__global__ __launch_bounds__(256)
void moe_expert_fallback(const float* __restrict__ x,
                         const float* __restrict__ Wg,
                         const float* __restrict__ Wu,
                         const float* __restrict__ Wd,
                         const int*   __restrict__ tok_list,
                         const float* __restrict__ w_list,
                         const int*   __restrict__ counts,
                         const int*   __restrict__ offsets,
                         const int*   __restrict__ ntiles,
                         const int*   __restrict__ tile_e,
                         const int*   __restrict__ tile_r,
                         float* __restrict__ out) {
  __shared__ __bf16 s_hmid[TILE_M * HSTRIDE];
  __shared__ float  s_araw[TILE_M * 32];
  __shared__ __bf16 s_a [TILE_M * BSTR];
  __shared__ __bf16 s_b0[64 * BSTR];
  __shared__ __bf16 s_b1[64 * BSTR];
  __shared__ int    s_tok[TILE_M];
  __shared__ float  s_w  [TILE_M];

  int bid = blockIdx.x;
  if (bid >= *ntiles) return;
  int e    = tile_e[bid];
  int trow = tile_r[bid];
  int base = offsets[e] + trow * TILE_M;
  int rows = min(TILE_M, counts[e] - trow * TILE_M);

  int tid  = threadIdx.x;
  int lane = tid & 31;
  int wave = tid >> 5;
  int wm   = wave >> 2;
  int wn   = wave & 3;

  if (tid < TILE_M) {
    int idx = base + min(tid, rows - 1);
    s_tok[tid] = tok_list[idx];
    s_w[tid]   = (tid < rows) ? w_list[idx] : 0.0f;
  }
  __syncthreads();

  const int m   = lane & 15;
  const int kba = (lane < 16) ? 0 : 16;
  const int kbb = (lane < 16) ? 0 : 16;
  const size_t wBase = (size_t)e * II * HH;
  const size_t dBase = (size_t)e * HH * II;

  for (int ni = 0; ni < II; ni += 64) {
    v8f accg = {0.f,0.f,0.f,0.f,0.f,0.f,0.f,0.f};
    v8f accu = {0.f,0.f,0.f,0.f,0.f,0.f,0.f,0.f};
    for (int k = 0; k < HH; k += 32) {
      __syncthreads();
      { // async-copy A tile: 32 rows x 32 fp32, one B128 per thread
        int r  = tid >> 3;
        int c0 = (tid & 7) * 4;
        const float* src = x + (size_t)s_tok[r] * HH + k + c0;
        unsigned dst = (unsigned)(size_t)(&s_araw[r * 32 + c0]);
        asm volatile("global_load_async_to_lds_b128 %0, %1, off"
                     :: "v"(dst), "v"(src) : "memory");
      }
      { // stage gate/up weights
        int n   = tid >> 2;
        int kk0 = (tid & 3) * 8;
        const float* sg = Wg + wBase + (size_t)(ni + n) * HH + k + kk0;
        const float* su = Wu + wBase + (size_t)(ni + n) * HH + k + kk0;
#pragma unroll
        for (int j = 0; j < 8; ++j) {
          s_b0[n * BSTR + kk0 + j] = f2bf(sg[j]);
          s_b1[n * BSTR + kk0 + j] = f2bf(su[j]);
        }
      }
      asm volatile("s_wait_asynccnt 0" ::: "memory");
      __syncthreads();
      {
        int r  = tid >> 3;
        int c0 = (tid & 7) * 4;
#pragma unroll
        for (int j = 0; j < 4; ++j)
          s_a[r * BSTR + c0 + j] = f2bf(s_araw[r * 32 + c0 + j]);
      }
      __syncthreads();
      v16bf af = ld_frag2(&s_a[(wm * 16 + m) * BSTR + kba], 8);
      v16bf bg = ld_frag2(&s_b0[(wn * 16 + m) * BSTR + kbb], 8);
      v16bf bu = ld_frag2(&s_b1[(wn * 16 + m) * BSTR + kbb], 8);
      accg = __builtin_amdgcn_wmma_f32_16x16x32_bf16(false, af, false, bg,
                                                     (short)0, accg, false, false);
      accu = __builtin_amdgcn_wmma_f32_16x16x32_bf16(false, af, false, bu,
                                                     (short)0, accu, false, false);
    }
    {
      int mrow = (lane < 16) ? 0 : 8;
#pragma unroll
      for (int r = 0; r < 8; ++r) {
        float g   = accg[r];
        float u   = accu[r];
        float sig = __builtin_amdgcn_rcpf(1.0f + __expf(-g));
        float hm  = g * sig * u;
        s_hmid[(wm * 16 + mrow + r) * HSTRIDE + ni + wn * 16 + m] = f2bf(hm);
      }
    }
  }
  __syncthreads();

  for (int n0 = 0; n0 < HH; n0 += 64) {
    v8f acc = {0.f,0.f,0.f,0.f,0.f,0.f,0.f,0.f};
    for (int k = 0; k < II; k += 32) {
      __syncthreads();
      {
        int n   = tid >> 2;
        int kk0 = (tid & 3) * 8;
        const float* sd = Wd + dBase + (size_t)(n0 + n) * II + k + kk0;
#pragma unroll
        for (int j = 0; j < 8; ++j) s_b0[n * BSTR + kk0 + j] = f2bf(sd[j]);
      }
      __syncthreads();
      v16bf af = ld_frag2(&s_hmid[(wm * 16 + m) * HSTRIDE + k + kba], 8);
      v16bf bd = ld_frag2(&s_b0[(wn * 16 + m) * BSTR + kbb], 8);
      acc = __builtin_amdgcn_wmma_f32_16x16x32_bf16(false, af, false, bd,
                                                    (short)0, acc, false, false);
    }
    {
      int mrow = (lane < 16) ? 0 : 8;
      int col  = n0 + wn * 16 + m;
#pragma unroll
      for (int r = 0; r < 8; ++r) {
        int row   = wm * 16 + mrow + r;
        float val = acc[r] * s_w[row];
        atomicAdd(&out[(size_t)s_tok[row] * HH + col], val);
      }
    }
  }
}

// ---------------- host launcher ----------------
extern "C" void kernel_launch(void* const* d_in, const int* in_sizes, int n_in,
                              void* d_out, int out_size, void* d_ws, size_t ws_size,
                              hipStream_t stream) {
  const float* x   = (const float*)d_in[0];   // hidden_states (B,S,H)
  const float* Wg  = (const float*)d_in[1];   // gate_weight (E,I,H)
  const float* Wu  = (const float*)d_in[2];   // up_weight   (E,I,H)
  const float* Wd  = (const float*)d_in[3];   // down_weight (E,H,I)
  const int*   idx = (const int*)d_in[4];     // topk_idx (N,K)
  const float* tw  = (const float*)d_in[5];   // topk_weight (N,K)
  float* out = (float*)d_out;

  // ---- workspace carve-up: 1 MB routing meta, then bf16 banks ----
  int* counts   = (int*)d_ws;                  // E
  int* cursors  = counts + EE;                 // E
  int* offsets  = cursors + EE;                // E+1
  int* ntiles   = offsets + EE + 1;            // 1
  int* tile_e   = ntiles + 1;                  // MAX_TILES
  int* tile_r   = tile_e + MAX_TILES;          // MAX_TILES
  int* tok_list = tile_r + MAX_TILES;          // AT
  float* w_list = (float*)(tok_list + AT);     // AT

  const size_t META  = 1u << 20;
  const size_t XBF_B = (size_t)NN * HH * 2;             // 64 MB
  const size_t PK_B  = (size_t)EE * II * HH * 2;        // 128 MB per bank
  const size_t need  = META + XBF_B + 3 * PK_B;

  __bf16* xbf = (__bf16*)((char*)d_ws + META);
  __bf16* pkg = (__bf16*)((char*)d_ws + META + XBF_B);
  __bf16* pku = (__bf16*)((char*)d_ws + META + XBF_B + PK_B);
  __bf16* pkd = (__bf16*)((char*)d_ws + META + XBF_B + 2 * PK_B);

  size_t nOut = (size_t)NN * HH;

  moe_zero_kernel<<<4096, 256, 0, stream>>>(out, nOut, counts, 2 * EE);
  moe_count_kernel<<<(AT + 255) / 256, 256, 0, stream>>>(idx, counts);
  moe_scan_kernel<<<1, 32, 0, stream>>>(counts, offsets, ntiles, tile_e, tile_r);
  moe_scatter_kernel<<<(AT + 255) / 256, 256, 0, stream>>>(idx, tw, offsets, cursors,
                                                           tok_list, w_list);

  if (ws_size >= need) {
    moe_cvt_x_kernel<<<4096, 256, 0, stream>>>(x, xbf, nOut);
    dim3 pgrid(EE * 16 * 64, 3);               // 32768 tiles per bank
    moe_pack_kernel<<<pgrid, 256, 0, stream>>>(Wg, Wu, Wd, pkg, pku, pkd);
    moe_expert_fast<<<MAX_TILES, 256, 0, stream>>>(xbf, pkg, pku, pkd,
                                                   tok_list, w_list,
                                                   counts, offsets, ntiles,
                                                   tile_e, tile_r, out);
  } else {
    moe_expert_fallback<<<MAX_TILES, 256, 0, stream>>>(x, Wg, Wu, Wd,
                                                       tok_list, w_list,
                                                       counts, offsets, ntiles,
                                                       tile_e, tile_r, out);
  }
}
